// MoCAttention_17583596110239
// MI455X (gfx1250) — matrix operand order repaired and checked by reading, not compiled, not run on hardware
//
#include <hip/hip_runtime.h>

typedef __attribute__((ext_vector_type(16))) __bf16 v16bf;
typedef __attribute__((ext_vector_type(8)))  __bf16 v8bf;
typedef __attribute__((ext_vector_type(8)))  float  v8f;
typedef __attribute__((ext_vector_type(4)))  unsigned int u32x4;
typedef __attribute__((ext_vector_type(4)))  int i32x4;
typedef __attribute__((ext_vector_type(8)))  int i32x8;

#define SEQ      2048
#define DMODEL   512
#define NHEAD    8
#define HDIM     64
#define NCHUNK   32
#define CHUNKLEN 64
#define TOPK     5

#if defined(__has_builtin)
#  if __has_builtin(__builtin_amdgcn_tensor_load_to_lds) && \
      __has_builtin(__builtin_amdgcn_s_wait_tensorcnt)
#    define HAVE_TDM 1
#  endif
#endif
#ifndef HAVE_TDM
#  define HAVE_TDM 0
#endif

static __device__ __forceinline__ v16bf pack16(v8bf lo, v8bf hi) {
  return __builtin_shufflevector(lo, hi, 0,1,2,3,4,5,6,7,8,9,10,11,12,13,14,15);
}
// A-fragment per CDNA5 layout: lane holds row M=lane&15; elems 0..7 at k=kb..kb+7,
// elems 8..15 at k=kb+16..kb+23 (kb = (lane>>4)*8). Caller passes p = &row[kb].
static __device__ __forceinline__ v16bf load_a_frag(const __bf16* p) {
  return pack16(*(const v8bf*)p, *(const v8bf*)(p + 16));
}
// B-fragment: lane holds column N=lane&15; elems 0..7 at k=kb..kb+7, elems 8..15 at
// k=kb+16..kb+23 (kb = (lane>>4)*8). Caller passes p = &B[kb][n], ld = row stride.
static __device__ __forceinline__ v16bf load_b_frag(const __bf16* p, int ld) {
  v16bf r;
  #pragma unroll
  for (int s = 0; s < 8; ++s) { r[s] = p[s * ld]; r[8 + s] = p[(16 + s) * ld]; }
  return r;
}

#if HAVE_TDM
// Issue a TDM 2-D tile load (bf16 elements) global -> LDS.
// D# packing per CDNA5 ISA 8.3/8.4: group0 = {count/type/lds_addr/global_addr},
// group1 = {data_size, tensor dims, tile dims, dim0 stride}. Dims in elements.
static __device__ __forceinline__ void tdm_load_2d_bf16(unsigned lds_off, const void* gptr,
                                                        unsigned tensor_d0, unsigned tensor_d1,
                                                        unsigned tile_d0, unsigned tile_d1,
                                                        unsigned stride0_elems) {
  unsigned long long ga = (unsigned long long)gptr;
  u32x4 g0;
  g0[0] = 1u;                                     // count=1, user descriptor
  g0[1] = lds_off;                                // LDS byte address
  g0[2] = (unsigned)ga;                           // global_addr[31:0]
  g0[3] = (unsigned)(ga >> 32) | 0x80000000u;     // global_addr[56:32] | type=2<<30
  i32x8 g1;
  g1[0] = (int)(1u << 16);                        // workgroup_mask=0, data_size=1 (2B)
  g1[1] = (int)((tensor_d0 & 0xFFFFu) << 16);     // tensor_dim0[15:0] at bits 63:48
  g1[2] = (int)(((tensor_d0 >> 16) & 0xFFFFu) | ((tensor_d1 & 0xFFFFu) << 16));
  g1[3] = (int)(((tensor_d1 >> 16) & 0xFFFFu) | (tile_d0 << 16));
  g1[4] = (int)tile_d1;                           // tile_dim1, tile_dim2=0
  g1[5] = (int)stride0_elems;                     // tensor_dim0_stride[31:0]
  g1[6] = 0;                                      // stride0[47:32]=0, stride1 lo=0
  g1[7] = 0;
  i32x4 gz4 = {0, 0, 0, 0};
#if __clang_major__ >= 23
  i32x8 gz8 = {0, 0, 0, 0, 0, 0, 0, 0};
  __builtin_amdgcn_tensor_load_to_lds(g0, g1, gz4, gz4, gz8, 0);
#else
  __builtin_amdgcn_tensor_load_to_lds(g0, g1, gz4, gz4, 0);
#endif
}
static __device__ __forceinline__ unsigned lds_byte_off(const void* p) {
  return (unsigned)(unsigned long long)p;         // LDS generic addr: low 32 bits = offset
}
#endif

// ---------------------------------------------------------------- fp32 -> bf16
__global__ void cvt_f32_bf16(const float* __restrict__ src, __bf16* __restrict__ dst, int n) {
  int i = blockIdx.x * blockDim.x + threadIdx.x;
  if (i < n) dst[i] = (__bf16)src[i];
}

// ---------------------------------------------------------------- tiled bf16 WMMA GEMM
// C[M,N] = scale * (A[M,K] @ B[K,N]); block tile 128x64, 8 waves (4x2) of 32x32.
// TDM double-buffered: wave 0 DMA-stages the next K-tile while all waves run WMMA.
template <typename OutT>
__global__ __launch_bounds__(256) void gemm_bf16(const __bf16* __restrict__ A,
                                                 const __bf16* __restrict__ B,
                                                 OutT* __restrict__ C,
                                                 int M, int N, int K, float scale) {
  __shared__ __bf16 As[2][128 * 32];
  __shared__ __bf16 Bs[2][32 * 64];
  const int tid  = threadIdx.x;
  const int lane = tid & 31;
  const int wid  = tid >> 5;
  const int wm   = (wid & 3) * 32;
  const int wn   = (wid >> 2) * 32;
  const int m0   = blockIdx.y * 128;
  const int n0   = blockIdx.x * 64;
  const int rh   = lane >> 4;
  const int cl   = lane & 15;
  const int nk   = K >> 5;

  v8f acc[2][2] = {};

#if HAVE_TDM
  if (wid == 0) {  // prologue: stage tile 0
    tdm_load_2d_bf16(lds_byte_off(&As[0][0]), &A[(size_t)m0 * K], K, 128, 32, 128, K);
    tdm_load_2d_bf16(lds_byte_off(&Bs[0][0]), &B[n0], N, 32, 64, 32, N);
  }
  for (int i = 0; i < nk; ++i) {
    const int cur = i & 1;
    if (wid == 0) __builtin_amdgcn_s_wait_tensorcnt(0);   // tile i landed
    __syncthreads();                                      // all waves done with buf cur^1
    if (wid == 0 && i + 1 < nk) {                         // overlap next DMA with compute
      const int k0n = (i + 1) * 32;
      tdm_load_2d_bf16(lds_byte_off(&As[cur ^ 1][0]), &A[(size_t)m0 * K + k0n], K, 128, 32, 128, K);
      tdm_load_2d_bf16(lds_byte_off(&Bs[cur ^ 1][0]), &B[(size_t)k0n * N + n0], N, 32, 64, 32, N);
    }
    #pragma unroll
    for (int ii = 0; ii < 2; ++ii) {
      v16bf a = load_a_frag(&As[cur][(wm + ii * 16 + cl) * 32 + rh * 8]);
      #pragma unroll
      for (int j = 0; j < 2; ++j) {
        v16bf b = load_b_frag(&Bs[cur][rh * 8 * 64 + wn + j * 16 + cl], 64);
        acc[ii][j] = __builtin_amdgcn_wmma_f32_16x16x32_bf16(
            false, a, false, b, (short)0, acc[ii][j], false, false);
      }
    }
  }
#else
  for (int i = 0; i < nk; ++i) {
    const int k0 = i * 32, cur = i & 1;
    {
      int c = tid * 2;
      #pragma unroll
      for (int ii = 0; ii < 2; ++ii, ++c) {
        int row = c >> 2, k8 = (c & 3) * 8;
        *(v8bf*)&As[cur][row * 32 + k8] = *(const v8bf*)&A[(size_t)(m0 + row) * K + k0 + k8];
      }
      int row = tid >> 3, n8 = (tid & 7) * 8;
      *(v8bf*)&Bs[cur][row * 64 + n8] = *(const v8bf*)&B[(size_t)(k0 + row) * N + n0 + n8];
    }
    __syncthreads();
    #pragma unroll
    for (int ii = 0; ii < 2; ++ii) {
      v16bf a = load_a_frag(&As[cur][(wm + ii * 16 + cl) * 32 + rh * 8]);
      #pragma unroll
      for (int j = 0; j < 2; ++j) {
        v16bf b = load_b_frag(&Bs[cur][rh * 8 * 64 + wn + j * 16 + cl], 64);
        acc[ii][j] = __builtin_amdgcn_wmma_f32_16x16x32_bf16(
            false, a, false, b, (short)0, acc[ii][j], false, false);
      }
    }
    __syncthreads();
  }
#endif

  #pragma unroll
  for (int i = 0; i < 2; ++i)
    #pragma unroll
    for (int j = 0; j < 2; ++j)
      #pragma unroll
      for (int r = 0; r < 8; ++r) {
        int row = m0 + wm + i * 16 + r + 8 * rh;
        int col = n0 + wn + j * 16 + cl;
        C[(size_t)row * N + col] = (OutT)(acc[i][j][r] * scale);
      }
}

// ---------------------------------------------------------------- chunk descriptors
__global__ void chunk_keys_kernel(const __bf16* __restrict__ Kb, float* __restrict__ ck) {
  int h = blockIdx.x >> 5, c = blockIdx.x & 31, d = threadIdx.x;
  float s = 0.f;
  #pragma unroll 8
  for (int r = 0; r < CHUNKLEN; ++r)
    s += (float)Kb[(size_t)(c * CHUNKLEN + r) * DMODEL + h * HDIM + d];
  ck[(h * NCHUNK + c) * HDIM + d] = s * (1.f / CHUNKLEN);
}

// ---------------------------------------------------------------- top-5 routing
__global__ __launch_bounds__(256) void routing_kernel(const __bf16* __restrict__ Qb,
                                                      const float* __restrict__ ck,
                                                      int* __restrict__ topk) {
  int t = blockIdx.x * blockDim.x + threadIdx.x;
  if (t >= NHEAD * SEQ) return;
  int h = t >> 11, s = t & (SEQ - 1);
  float q[HDIM];
  #pragma unroll
  for (int d = 0; d < HDIM; ++d)
    q[d] = (float)Qb[(size_t)s * DMODEL + h * HDIM + d];   // Q already pre-scaled
  float sims[NCHUNK];
  for (int c = 0; c < NCHUNK; ++c) {
    const float* ckp = &ck[(h * NCHUNK + c) * HDIM];
    float acc = 0.f;
    #pragma unroll
    for (int d = 0; d < HDIM; ++d) acc += q[d] * ckp[d];
    sims[c] = acc;
  }
  unsigned chosen = 0;
  #pragma unroll
  for (int k = 0; k < TOPK; ++k) {
    float best = -3.0e38f; int bi = 0;
    #pragma unroll
    for (int c = 0; c < NCHUNK; ++c)
      if (!((chosen >> c) & 1u) && sims[c] > best) { best = sims[c]; bi = c; }
    chosen |= 1u << bi;
    topk[(size_t)t * TOPK + k] = bi;
  }
}

// ---------------------------------------------------------------- masked flash attention
// Block = 16 queries x 1 head, 4 waves each owning chunks {wave, wave+4, ...} of the
// union of selected chunks; per-row top-k mask gives exact sparse semantics.
__global__ __launch_bounds__(128) void moc_attn_kernel(const __bf16* __restrict__ Qb,
                                                       const __bf16* __restrict__ Kb,
                                                       const __bf16* __restrict__ Vb,
                                                       const int* __restrict__ topk,
                                                       __bf16* __restrict__ Ob) {
  __shared__ __bf16 Kt[4][CHUNKLEN * HDIM];   // [wave][d*64 + n]  (K chunk, transposed)
  __shared__ __bf16 Pc[4][16 * CHUNKLEN];     // [wave][m*64 + n]  (exp probs)
  __shared__ float  Mw[4][16], Lw[4][16];
  __shared__ float  Osum[16 * HDIM];
  __shared__ float  Linv[16];
  __shared__ unsigned rowmask[16];
  __shared__ unsigned unionmask;

  const int h    = blockIdx.y;
  const int q0   = blockIdx.x * 16;
  const int tid  = threadIdx.x;
  const int wave = tid >> 5;
  const int lane = tid & 31;
  const int rh   = lane >> 4;
  const int cl   = lane & 15;

  if (tid == 0) unionmask = 0;
  for (int i = tid; i < 16 * HDIM; i += 128) Osum[i] = 0.f;
  __syncthreads();
  if (tid < 16) {
    unsigned m = 0;
    #pragma unroll
    for (int k = 0; k < TOPK; ++k)
      m |= 1u << topk[(size_t)(h * SEQ + q0 + tid) * TOPK + k];
    rowmask[tid] = m;
    atomicOr(&unionmask, m);
  }
  __syncthreads();
  const unsigned um = unionmask;

  unsigned rmask[8];
  #pragma unroll
  for (int r = 0; r < 8; ++r) rmask[r] = rowmask[r + 8 * rh];

  v16bf qf[2];
  #pragma unroll
  for (int ks = 0; ks < 2; ++ks)
    qf[ks] = load_a_frag(&Qb[(size_t)(q0 + cl) * DMODEL + h * HDIM + ks * 32 + rh * 8]);

  v8f oacc[4] = {};
  float mrow[8], lrow[8];
  #pragma unroll
  for (int r = 0; r < 8; ++r) { mrow[r] = -3.0e38f; lrow[r] = 0.f; }

  for (int c = wave; c < NCHUNK; c += 4) {           // uniform per wave -> EXEC stays full
    if (!((um >> c) & 1u)) continue;

    // stage K chunk transposed into this wave's LDS slab
    #pragma unroll
    for (int nn = 0; nn < 2; ++nn) {
      int n = lane + nn * 32;
      const __bf16* src = &Kb[(size_t)(c * CHUNKLEN + n) * DMODEL + h * HDIM];
      #pragma unroll 8
      for (int d = 0; d < HDIM; ++d)
        Kt[wave][d * CHUNKLEN + n] = src[d];
    }
    asm volatile("s_wait_dscnt 0" ::: "memory");     // same-wave LDS store->load ordering

    // S = Q @ K^T  (16 x 64, fp32 accum)
    v8f sacc[4];
    #pragma unroll
    for (int nt = 0; nt < 4; ++nt) {
      v8f a = {};
      #pragma unroll
      for (int ks = 0; ks < 2; ++ks) {
        v16bf b = load_b_frag(&Kt[wave][(ks * 32 + rh * 8) * CHUNKLEN + nt * 16 + cl], CHUNKLEN);
        a = __builtin_amdgcn_wmma_f32_16x16x32_bf16(false, qf[ks], false, b, (short)0, a,
                                                    false, false);
      }
      sacc[nt] = a;
    }

    // masked online-softmax row update (row = r + 8*rh, col = cl + 16*nt)
    #pragma unroll
    for (int r = 0; r < 8; ++r) {
      const bool sel = (rmask[r] >> c) & 1u;
      float v = sel ? fmaxf(fmaxf(sacc[0][r], sacc[1][r]), fmaxf(sacc[2][r], sacc[3][r]))
                    : -3.0e38f;
      #pragma unroll
      for (int off = 1; off < 16; off <<= 1) v = fmaxf(v, __shfl_xor(v, off, 32));
      float mn = fmaxf(mrow[r], v);
      float rescale = __expf(mrow[r] - mn);
      mrow[r] = mn;
      lrow[r] *= rescale;
      #pragma unroll
      for (int nt = 0; nt < 4; ++nt) oacc[nt][r] *= rescale;
      float psum = 0.f;
      #pragma unroll
      for (int nt = 0; nt < 4; ++nt) {
        float e = sel ? __expf(sacc[nt][r] - mn) : 0.f;
        Pc[wave][(r + 8 * rh) * CHUNKLEN + nt * 16 + cl] = (__bf16)e;
        psum += e;
      }
      #pragma unroll
      for (int off = 1; off < 16; off <<= 1) psum += __shfl_xor(psum, off, 32);
      lrow[r] += psum;
    }
    asm volatile("s_wait_dscnt 0" ::: "memory");

    // O += P @ V  (V rows are already B-layout friendly: lane = feat, 16 keys strided by DMODEL)
    #pragma unroll
    for (int ks = 0; ks < 2; ++ks) {
      v16bf a = load_a_frag(&Pc[wave][cl * CHUNKLEN + ks * 32 + rh * 8]);
      #pragma unroll
      for (int nt = 0; nt < 4; ++nt) {
        v16bf b = load_b_frag(&Vb[(size_t)(c * CHUNKLEN + ks * 32 + rh * 8) * DMODEL +
                                  h * HDIM + nt * 16 + cl], DMODEL);
        oacc[nt] = __builtin_amdgcn_wmma_f32_16x16x32_bf16(false, a, false, b, (short)0,
                                                           oacc[nt], false, false);
      }
    }
  }

  // flash-style merge of the 4 waves' partials
  if (cl == 0) {
    #pragma unroll
    for (int r = 0; r < 8; ++r) {
      Mw[wave][r + 8 * rh] = mrow[r];
      Lw[wave][r + 8 * rh] = lrow[r];
    }
  }
  __syncthreads();
  #pragma unroll
  for (int r = 0; r < 8; ++r) {
    int row = r + 8 * rh;
    float gm = -3.0e38f;
    #pragma unroll
    for (int w = 0; w < 4; ++w) gm = fmaxf(gm, Mw[w][row]);
    float gl = 0.f;
    #pragma unroll
    for (int w = 0; w < 4; ++w) gl += Lw[w][row] * __expf(Mw[w][row] - gm);
    float f = __expf(mrow[r] - gm);
    #pragma unroll
    for (int nt = 0; nt < 4; ++nt)
      atomicAdd(&Osum[row * HDIM + nt * 16 + cl], oacc[nt][r] * f);
    if (wave == 0 && cl == 0) Linv[row] = 1.f / gl;
  }
  __syncthreads();
  for (int i = tid; i < 16 * HDIM; i += 128) {
    int row = i >> 6, d = i & 63;
    Ob[(size_t)(q0 + row) * DMODEL + h * HDIM + d] = (__bf16)(Osum[i] * Linv[row]);
  }
}

// ---------------------------------------------------------------- launch
extern "C" void kernel_launch(void* const* d_in, const int* in_sizes, int n_in,
                              void* d_out, int out_size, void* d_ws, size_t ws_size,
                              hipStream_t stream) {
  const float* x  = (const float*)d_in[0];
  const float* Wq = (const float*)d_in[1];
  const float* Wk = (const float*)d_in[2];
  const float* Wv = (const float*)d_in[3];
  const float* Wo = (const float*)d_in[4];

  char* p = (char*)d_ws;
  auto take = [&](size_t bytes) { char* r = p; p += (bytes + 255) & ~(size_t)255; return r; };
  __bf16* Xb  = (__bf16*)take((size_t)SEQ * DMODEL * 2);
  __bf16* Wqb = (__bf16*)take((size_t)DMODEL * DMODEL * 2);
  __bf16* Wkb = (__bf16*)take((size_t)DMODEL * DMODEL * 2);
  __bf16* Wvb = (__bf16*)take((size_t)DMODEL * DMODEL * 2);
  __bf16* Wob = (__bf16*)take((size_t)DMODEL * DMODEL * 2);
  __bf16* Qb  = (__bf16*)take((size_t)SEQ * DMODEL * 2);
  __bf16* Kb  = (__bf16*)take((size_t)SEQ * DMODEL * 2);
  __bf16* Vb  = (__bf16*)take((size_t)SEQ * DMODEL * 2);
  __bf16* Ob  = (__bf16*)take((size_t)SEQ * DMODEL * 2);
  float*  ck  = (float*)take((size_t)NHEAD * NCHUNK * HDIM * 4);
  int*    tk  = (int*)take((size_t)NHEAD * SEQ * TOPK * 4);

  const int nX = SEQ * DMODEL;
  const int nW = DMODEL * DMODEL;
  cvt_f32_bf16<<<(nX + 255) / 256, 256, 0, stream>>>(x, Xb, nX);
  cvt_f32_bf16<<<(nW + 255) / 256, 256, 0, stream>>>(Wq, Wqb, nW);
  cvt_f32_bf16<<<(nW + 255) / 256, 256, 0, stream>>>(Wk, Wkb, nW);
  cvt_f32_bf16<<<(nW + 255) / 256, 256, 0, stream>>>(Wv, Wvb, nW);
  cvt_f32_bf16<<<(nW + 255) / 256, 256, 0, stream>>>(Wo, Wob, nW);

  dim3 ggrid(DMODEL / 64, SEQ / 128);
  // Q pre-scaled by hd^-0.5 = 0.125 so routing sims and attention scores inherit the scale
  gemm_bf16<__bf16><<<ggrid, 256, 0, stream>>>(Xb, Wqb, Qb, SEQ, DMODEL, DMODEL, 0.125f);
  gemm_bf16<__bf16><<<ggrid, 256, 0, stream>>>(Xb, Wkb, Kb, SEQ, DMODEL, DMODEL, 1.0f);
  gemm_bf16<__bf16><<<ggrid, 256, 0, stream>>>(Xb, Wvb, Vb, SEQ, DMODEL, DMODEL, 1.0f);

  chunk_keys_kernel<<<NHEAD * NCHUNK, 64, 0, stream>>>(Kb, ck);
  routing_kernel<<<(NHEAD * SEQ) / 256, 256, 0, stream>>>(Qb, ck, tk);
  moc_attn_kernel<<<dim3(SEQ / 16, NHEAD), 128, 0, stream>>>(Qb, Kb, Vb, tk, Ob);

  gemm_bf16<float><<<ggrid, 256, 0, stream>>>(Ob, Wob, (float*)d_out, SEQ, DMODEL, DMODEL, 1.0f);
}
